// WaveletNauralNet_78769700208780
// MI455X (gfx1250) — compile-verified, matrix operand order checked by reading, not run on hardware
//
#include <hip/hip_runtime.h>
#include <hip/hip_bf16.h>

// ---------------------------------------------------------------------------
// WaveletNeuralNet for MI455X (gfx1250, wave32, WMMA + TDM)
//   K0: split W0 into bf16 hi/lo planes, N padded 181->192 with zeros.
//   K1: per-row db8 wavelet pyramid + reconstructions, LDS-resident,
//       writes relu(features) as bf16 hi/lo planes (2048 x 32768 each).
//   K2: GEMM relu(feat) @ W0^T via v_wmma_f32_16x16x32_bf16, bf16x3 split,
//       tiles staged by the Tensor Data Mover (tensor_load_to_lds) with
//       double-buffered LDS and s_wait_tensorcnt; K split x8, atomic f32
//       reduction into C1.
//   K3: tail MLP 181 -> 13 -> 10.
// ---------------------------------------------------------------------------

typedef __attribute__((ext_vector_type(16))) __bf16    v16bf;
typedef __attribute__((ext_vector_type(8)))  float     v8f;
typedef __attribute__((ext_vector_type(4)))  unsigned  u32x4;
typedef __attribute__((ext_vector_type(8)))  int       i32x8;
typedef __attribute__((ext_vector_type(4)))  int       i32x4;

#define BATCH 2048
#define NSIG  4096
#define DIN   32768
#define H1D   181
#define NPAD  192
#define H2D   13
#define OUTD  10

#if __has_builtin(__builtin_amdgcn_tensor_load_to_lds) && \
    __has_builtin(__builtin_amdgcn_s_wait_tensorcnt)
#define USE_TDM 1
#else
#define USE_TDM 0
#endif

__device__ __constant__ float C_LO[8] = {
     0.23037781330885523f,  0.7148465705525415f,   0.6308807679295904f,
    -0.02798376941698385f, -0.18703481171888114f,  0.030841381835986965f,
     0.032883011666982945f,-0.010597401784997278f };
__device__ __constant__ float C_HI[8] = {
    -0.010597401784997278f,-0.032883011666982945f, 0.030841381835986965f,
     0.18703481171888114f, -0.02798376941698385f, -0.6308807679295904f,
     0.7148465705525415f,  -0.23037781330885523f };

__device__ __forceinline__ __bf16 f2bf(float f) {
    unsigned u = __builtin_bit_cast(unsigned, f);
    unsigned r = (u + 0x7FFFu + ((u >> 16) & 1u)) >> 16;   // RNE
    unsigned short s = (unsigned short)r;
    return __builtin_bit_cast(__bf16, s);
}
__device__ __forceinline__ float bf2f(__bf16 b) {
    unsigned short s = __builtin_bit_cast(unsigned short, b);
    unsigned u = ((unsigned)s) << 16;
    return __builtin_bit_cast(float, u);
}

// ------------------------------- Kernel 0 ----------------------------------
// W0 (181 x 32768) -> bf16 hi/lo planes (192 x 32768), rows 181..191 zeroed.
__global__ __launch_bounds__(256) void prep_w0_kernel(const float* __restrict__ W0,
                                                      __bf16* __restrict__ w0hi,
                                                      __bf16* __restrict__ w0lo)
{
    const size_t total  = (size_t)NPAD * DIN;
    const size_t stride = (size_t)gridDim.x * blockDim.x;
    for (size_t idx = (size_t)blockIdx.x * blockDim.x + threadIdx.x;
         idx < total; idx += stride) {
        int n = (int)(idx >> 15);
        int k = (int)(idx & (DIN - 1));
        float v = (n < H1D) ? W0[(size_t)n * DIN + k] : 0.f;
        __bf16 h = f2bf(v);
        w0hi[idx] = h;
        w0lo[idx] = f2bf(v - bf2f(h));
    }
}

// ------------------------------- Kernel 1 ----------------------------------
__global__ __launch_bounds__(256) void wavelet_kernel(const float* __restrict__ x,
                                                      __bf16* __restrict__ featHi,
                                                      __bf16* __restrict__ featLo)
{
    // LDS map (floats): XA[0,4096) XB[4096,6147) DD[6147,10285)
    //                   RB[10285,14381) filters[14381,14397); RA aliases XA.
    __shared__ float S[14400];
    const int tid = threadIdx.x;
    const int row = blockIdx.x;

    const int NL[10]  = {4096,2051,1029,518,262,134,70,38,22,14};
    const int DOFF[9] = {0,2051,3080,3598,3860,3994,4064,4102,4124};

    float* XA  = S;
    float* XB  = S + 4096;
    float* DD  = S + 6147;
    float* RA  = S;
    float* RB  = S + 10285;
    float* FLO = S + 14381;
    float* FHI = S + 14389;

    if (tid < 8) { FLO[tid] = C_LO[tid]; FHI[tid] = C_HI[tid]; }
    for (int j = tid; j < NSIG; j += 256) XA[j] = x[(size_t)row * NSIG + j];
    __syncthreads();

    // DWT cascade
    float* cur = XA;
    float* nxt = XB;
    for (int lev = 1; lev <= 9; ++lev) {
        const int n = NL[lev - 1];
        const int m = NL[lev];
        float* d = DD + DOFF[lev - 1];
        for (int k = tid; k < m; k += 256) {
            float lo = 0.f, hi = 0.f;
            #pragma unroll
            for (int t = 0; t < 8; ++t) {
                int i = 2 * k + t;
                int src;
                if (i < 6) src = 5 - i;
                else { i -= 6; src = (i < n) ? i : (2 * n - 1 - i); }
                float v = cur[src];
                lo += v * C_LO[t];
                hi += v * C_HI[t];
            }
            nxt[k] = lo;
            d[k]   = hi;
        }
        __syncthreads();
        float* t_ = cur; cur = nxt; nxt = t_;
    }

    // reconstructions for levels 2..9
    for (int lev = 2; lev <= 9; ++lev) {
        const int nl = NL[lev];
        const float* dl = DD + DOFF[lev - 1];
        float* rc = RA;
        int olen = 2 * nl - 6;
        for (int j = tid; j < olen; j += 256) {
            float acc = 0.f;
            int m0 = j >> 1;
            int m1 = (j + 6) >> 1; if (m1 > nl - 1) m1 = nl - 1;
            for (int mm = m0; mm <= m1; ++mm)
                acc += dl[mm] * FHI[j + 6 - 2 * mm];
            rc[j] = acc;
        }
        __syncthreads();

        for (int J = lev - 1; J >= 1; --J) {
            const int nJ = NL[J];
            const float* dj = (J == lev - 1) ? (DD + DOFF[J - 1]) : nullptr;
            float* out = (rc == RA) ? RB : RA;
            olen = 2 * nJ - 6;
            for (int j = tid; j < olen; j += 256) {
                float acc = 0.f;
                int m0 = j >> 1;
                int m1 = (j + 6) >> 1; if (m1 > nJ - 1) m1 = nJ - 1;
                if (dj) {
                    for (int mm = m0; mm <= m1; ++mm) {
                        int t = j + 6 - 2 * mm;
                        acc += rc[mm] * FLO[t] + dj[mm] * FHI[t];
                    }
                } else {
                    for (int mm = m0; mm <= m1; ++mm)
                        acc += rc[mm] * FLO[j + 6 - 2 * mm];
                }
                out[j] = acc;
            }
            __syncthreads();
            rc = out;
        }
        // fused ReLU + bf16 hi/lo split store
        size_t base = (size_t)row * DIN + (size_t)(lev - 2) * 4096;
        for (int j = tid; j < 4096; j += 256) {
            float v = fmaxf(rc[j], 0.f);
            __bf16 h = f2bf(v);
            featHi[base + j] = h;
            featLo[base + j] = f2bf(v - bf2f(h));
        }
        __syncthreads();
    }
}

// ------------------------------- Kernel 2 ----------------------------------
#if USE_TDM
// Issue one 2-D TDM tile load: rows x 32 bf16 elements, row stride 32768.
__device__ __forceinline__ void tdm_tile_load(unsigned lds_addr, const __bf16* gsrc,
                                              int rows)
{
    unsigned long long ga = (unsigned long long)(size_t)gsrc;
    u32x4 g0;
    g0[0] = 1u;                                            // count=1 (valid D#)
    g0[1] = lds_addr;                                      // LDS byte address
    g0[2] = (unsigned)(ga & 0xFFFFFFFFu);                  // global_addr[31:0]
    g0[3] = (unsigned)((ga >> 32) & 0x01FFFFFFu)           // global_addr[56:32]
          | (2u << 30);                                    // type = 2 ("image")
    i32x8 g1;
    g1[0] = 0x00010000;                                    // data_size=1 (2B)
    g1[1] = (int)(0x8000u << 16);                          // tensor_dim0=32768 lo16
    g1[2] = 0;                                             // dim0 hi16 / dim1 lo16
    g1[3] = (int)(1u | (32u << 16));                       // dim1 hi16=1, tile_dim0=32
    g1[4] = rows;                                          // tile_dim1, tile_dim2=0
    g1[5] = 32768;                                         // tensor_dim0_stride lo32
    g1[6] = 0;                                             // stride0 hi / stride1 lo
    g1[7] = 0;
    i32x4 g2; g2[0] = 0; g2[1] = 0; g2[2] = 0; g2[3] = 0;
    i32x4 g3; g3[0] = 0; g3[1] = 0; g3[2] = 0; g3[3] = 0;
    i32x8 g4; g4[0] = 0; g4[1] = 0; g4[2] = 0; g4[3] = 0;
    g4[4] = 0; g4[5] = 0; g4[6] = 0; g4[7] = 0;
    // 6-arg form (clang-23 / therock-10.0 lane)
    __builtin_amdgcn_tensor_load_to_lds(g0, g1, g2, g3, g4, 0);
}
#endif

// grid (32 M-tiles of 64 rows, 8 K-chunks of 4096); block 256 = 8 wave32.
// wave w: M-strip (w>>1)*16, N-half (w&1)*96 (6 x 16 N-tiles); K step 32.
__global__ __launch_bounds__(256) void gemm1_kernel(const __bf16* __restrict__ featHi,
                                                    const __bf16* __restrict__ featLo,
                                                    const __bf16* __restrict__ w0Hi,
                                                    const __bf16* __restrict__ w0Lo,
                                                    float* __restrict__ C1)
{
    __shared__ __bf16 sAhi[2][64][32];
    __shared__ __bf16 sAlo[2][64][32];
    __shared__ __bf16 sBhi[2][NPAD][32];
    __shared__ __bf16 sBlo[2][NPAD][32];

    const int tid    = threadIdx.x;
    const int lane   = tid & 31;
    const int wave   = tid >> 5;
    const int half   = lane >> 4;
    const int lr     = lane & 15;
    const int mstrip = wave >> 1;
    const int nhalf  = wave & 1;
    const int M0     = blockIdx.x * 64;
    const int Kbase  = blockIdx.y * 4096;
    const int NSTEP  = 4096 / 32;

    v8f acc[6];
    #pragma unroll
    for (int t = 0; t < 6; ++t)
        #pragma unroll
        for (int i = 0; i < 8; ++i) acc[t][i] = 0.f;

#if USE_TDM
    // preload buffer 0 (wave 0 only; TENSORcnt is per-wave)
    if (tid < 32) {
        const int k0 = Kbase;
        tdm_tile_load((unsigned)(size_t)&sAhi[0][0][0], featHi + (size_t)M0 * DIN + k0, 64);
        tdm_tile_load((unsigned)(size_t)&sAlo[0][0][0], featLo + (size_t)M0 * DIN + k0, 64);
        tdm_tile_load((unsigned)(size_t)&sBhi[0][0][0], w0Hi + k0, NPAD);
        tdm_tile_load((unsigned)(size_t)&sBlo[0][0][0], w0Lo + k0, NPAD);
    }
#endif

    for (int step = 0; step < NSTEP; ++step) {
        const int cur = step & 1;
#if USE_TDM
        if (tid < 32) __builtin_amdgcn_s_wait_tensorcnt(0);   // buf[cur] landed
        __syncthreads();   // publishes buf[cur]; everyone done reading buf[cur^1]
        if (tid < 32 && step + 1 < NSTEP) {
            const int nb = cur ^ 1;
            const int k1 = Kbase + (step + 1) * 32;
            tdm_tile_load((unsigned)(size_t)&sAhi[nb][0][0], featHi + (size_t)M0 * DIN + k1, 64);
            tdm_tile_load((unsigned)(size_t)&sAlo[nb][0][0], featLo + (size_t)M0 * DIN + k1, 64);
            tdm_tile_load((unsigned)(size_t)&sBhi[nb][0][0], w0Hi + k1, NPAD);
            tdm_tile_load((unsigned)(size_t)&sBlo[nb][0][0], w0Lo + k1, NPAD);
        }
#else
        {   // synchronous fallback staging
            const int k0 = Kbase + step * 32;
            for (int e = tid; e < 64 * 32; e += 256) {
                int r = e >> 5, c = e & 31;
                sAhi[cur][r][c] = featHi[(size_t)(M0 + r) * DIN + (k0 + c)];
                sAlo[cur][r][c] = featLo[(size_t)(M0 + r) * DIN + (k0 + c)];
            }
            for (int e = tid; e < NPAD * 32; e += 256) {
                int nn = e >> 5, c = e & 31;
                sBhi[cur][nn][c] = w0Hi[(size_t)nn * DIN + (k0 + c)];
                sBlo[cur][nn][c] = w0Lo[(size_t)nn * DIN + (k0 + c)];
            }
            __syncthreads();
        }
#endif
        // A fragment (16-bit A 16x32 layout: lane half selects K groups)
        const int rowt = mstrip * 16 + lr;
        v16bf ahi, alo;
        #pragma unroll
        for (int i = 0; i < 16; ++i) {
            int kk = (i < 8) ? (half * 8 + i) : (16 + half * 8 + (i - 8));
            ahi[i] = sAhi[cur][rowt][kk];
            alo[i] = sAlo[cur][rowt][kk];
        }
        #pragma unroll
        for (int t = 0; t < 6; ++t) {
            int ncol = nhalf * 96 + t * 16 + lr;
            v16bf bhi, blo;
            #pragma unroll
            for (int i = 0; i < 16; ++i) {
                int kk = (i < 8) ? (half * 8 + i) : (16 + half * 8 + (i - 8));
                bhi[i] = sBhi[cur][ncol][kk];
                blo[i] = sBlo[cur][ncol][kk];
            }
            // bf16x3 split: hi*hi + hi*lo + lo*hi, f32 accumulate
            acc[t] = __builtin_amdgcn_wmma_f32_16x16x32_bf16(false, ahi, false, bhi,
                                                             (short)0, acc[t], false, false);
            acc[t] = __builtin_amdgcn_wmma_f32_16x16x32_bf16(false, ahi, false, blo,
                                                             (short)0, acc[t], false, false);
            acc[t] = __builtin_amdgcn_wmma_f32_16x16x32_bf16(false, alo, false, bhi,
                                                             (short)0, acc[t], false, false);
        }
#if !USE_TDM
        __syncthreads();
#endif
    }

    // epilogue: C/D layout — VGPR i: lanes0-15 M=i, lanes16-31 M=8+i; N=lane&15
    #pragma unroll
    for (int t = 0; t < 6; ++t) {
        int ncol = nhalf * 96 + t * 16 + lr;
        if (ncol < H1D) {
            #pragma unroll
            for (int i = 0; i < 8; ++i) {
                int m = M0 + mstrip * 16 + half * 8 + i;
                atomicAdd(&C1[(size_t)m * H1D + ncol], acc[t][i]);
            }
        }
    }
}

// ------------------------------- Kernel 3 ----------------------------------
__global__ __launch_bounds__(256) void tail_kernel(const float* __restrict__ C1,
                                                   const float* __restrict__ b0,
                                                   const float* __restrict__ W1,
                                                   const float* __restrict__ b1,
                                                   const float* __restrict__ W2,
                                                   const float* __restrict__ b2,
                                                   float* __restrict__ out)
{
    __shared__ float sW1[H2D * H1D];
    __shared__ float sW2[OUTD * H2D];
    __shared__ float sb0[H1D];
    __shared__ float sb1[H2D];
    __shared__ float sb2[OUTD];
    const int tid = threadIdx.x;
    for (int e = tid; e < H2D * H1D; e += 256) sW1[e] = W1[e];
    for (int e = tid; e < OUTD * H2D; e += 256) sW2[e] = W2[e];
    for (int e = tid; e < H1D; e += 256) sb0[e] = b0[e];
    if (tid < H2D)  sb1[tid] = b1[tid];
    if (tid < OUTD) sb2[tid] = b2[tid];
    __syncthreads();

    const int row = blockIdx.x * 256 + tid;
    const float* crow = C1 + (size_t)row * H1D;

    float a1[H2D];
    #pragma unroll
    for (int j = 0; j < H2D; ++j) a1[j] = sb1[j];
    for (int n = 0; n < H1D; ++n) {
        float v = fmaxf(crow[n] + sb0[n], 0.f);
        #pragma unroll
        for (int j = 0; j < H2D; ++j) a1[j] += v * sW1[j * H1D + n];
    }
    float a2[H2D];
    #pragma unroll
    for (int j = 0; j < H2D; ++j) a2[j] = fmaxf(a1[j], 0.f);
    #pragma unroll
    for (int o = 0; o < OUTD; ++o) {
        float acc = sb2[o];
        #pragma unroll
        for (int j = 0; j < H2D; ++j) acc += a2[j] * sW2[o * H2D + j];
        out[(size_t)row * OUTD + o] = acc;
    }
}

// ------------------------------- launcher ----------------------------------
extern "C" void kernel_launch(void* const* d_in, const int* in_sizes, int n_in,
                              void* d_out, int out_size, void* d_ws, size_t ws_size,
                              hipStream_t stream)
{
    const float* x1 = (const float*)d_in[0];
    const float* W0 = (const float*)d_in[3];
    const float* b0 = (const float*)d_in[4];
    const float* W1 = (const float*)d_in[5];
    const float* b1 = (const float*)d_in[6];
    const float* W2 = (const float*)d_in[7];
    const float* b2 = (const float*)d_in[8];
    float* out = (float*)d_out;

    // workspace layout
    char* ws = (char*)d_ws;
    const size_t featBytes = (size_t)BATCH * DIN * sizeof(__bf16);   // 128 MB each
    const size_t w0Bytes   = (size_t)NPAD * DIN * sizeof(__bf16);    // 12 MB each
    __bf16* featHi = (__bf16*)(ws);
    __bf16* featLo = (__bf16*)(ws + featBytes);
    __bf16* w0Hi   = (__bf16*)(ws + 2 * featBytes);
    __bf16* w0Lo   = (__bf16*)(ws + 2 * featBytes + w0Bytes);
    float*  C1     = (float*) (ws + 2 * featBytes + 2 * w0Bytes);

    (void)hipMemsetAsync(C1, 0, (size_t)BATCH * H1D * sizeof(float), stream);

    hipLaunchKernelGGL(prep_w0_kernel, dim3(512), dim3(256), 0, stream, W0, w0Hi, w0Lo);
    hipLaunchKernelGGL(wavelet_kernel, dim3(BATCH), dim3(256), 0, stream,
                       x1, featHi, featLo);
    hipLaunchKernelGGL(gemm1_kernel, dim3(BATCH / 64, 8), dim3(256), 0, stream,
                       featHi, featLo, w0Hi, w0Lo, C1);
    hipLaunchKernelGGL(tail_kernel, dim3(BATCH / 256), dim3(256), 0, stream,
                       C1, b0, W1, b1, W2, b2, out);
}
